// MoEMLP_10763188044537
// MI455X (gfx1250) — matrix operand order repaired and checked by reading, not compiled, not run on hardware
//
#include <hip/hip_runtime.h>
#include <hip/hip_bf16.h>

#define S_ 257
#define B_ 64
#define D_ 768
#define E_ 8
#define H_ 3072
#define T_ (S_ * B_)          // 16448 tokens == 257 * 64

#define HC 128                 // H chunk
#define XS_STRIDE 776          // bf16 elems/row (768 + 8 pad)
#define HS_STRIDE 136          // bf16 elems/row (128 + 8 pad)
#define YS_STRIDE 772          // f32  elems/row (768 + 4 pad)

typedef __attribute__((ext_vector_type(16))) __bf16 bf16x16;
typedef __attribute__((ext_vector_type(8)))  float  f32x8;

union FragAB { bf16x16 v; uint4 q[2]; };

#define WMMA_BF16(A, Bf, C) __builtin_amdgcn_wmma_f32_16x16x32_bf16( \
    false, (A).v, false, (Bf).v, (short)0, (C), false, false)

// ---- pack fp32 weights into bf16 WMMA B-fragments -------------------------------
// src: [E][NT*16][NK*32] f32 (row-major out,in) -> dst frags [E][NT][NK][32][16] bf16
// frag(e,nt,nk): lane = 16*kh + n holds B[k][n] for k = nk*32 + kh*16 + j (j=0..15),
// i.e. src row nt*16+n, 16 contiguous source floats -> 32B contiguous per lane.
__global__ __launch_bounds__(256) void pack_frags_kernel(const float* __restrict__ src,
                                                         __bf16* __restrict__ dst,
                                                         int NT, int NK) {
    int total = E_ * NT * NK * 32;
    int stride = gridDim.x * blockDim.x;
    for (int id = blockIdx.x * blockDim.x + threadIdx.x; id < total; id += stride) {
        int lane = id & 31;
        int f = id >> 5;
        int nk = f % NK;
        int nt = (f / NK) % NT;
        int e  = f / (NK * NT);
        const float* sp = src + ((size_t)(e * NT + nt) * 16 + (lane & 15)) * ((size_t)NK * 32)
                              + nk * 32 + (lane >> 4) * 16;
        union { __bf16 h[8]; uint4 u; } o0, o1;
#pragma unroll
        for (int j = 0; j < 8; ++j) { o0.h[j] = (__bf16)sp[j]; o1.h[j] = (__bf16)sp[8 + j]; }
        uint4* dp = (uint4*)(dst + (size_t)id * 16);
        dp[0] = o0.u; dp[1] = o1.u;
    }
}

// ---- router: logits = x @ Wr^T + br ; softmax over E ----------------------------
__global__ __launch_bounds__(256) void router_kernel(const float* __restrict__ x,
                                                     const float* __restrict__ Wr,
                                                     const float* __restrict__ br,
                                                     float* __restrict__ probs) {
    const int wave = threadIdx.x >> 5;
    const int lane = threadIdx.x & 31;
    const int t = blockIdx.x * 8 + wave;
    if (t >= T_) return;
    const float* xt = x + (size_t)t * D_;
    float acc[E_];
#pragma unroll
    for (int e = 0; e < E_; ++e) acc[e] = 0.f;
    for (int i = lane; i < D_; i += 32) {
        float xv = xt[i];
#pragma unroll
        for (int e = 0; e < E_; ++e) acc[e] += xv * Wr[e * D_ + i];
    }
#pragma unroll
    for (int e = 0; e < E_; ++e) {
        float v = acc[e];
#pragma unroll
        for (int off = 16; off > 0; off >>= 1) v += __shfl_xor(v, off, 32);
        acc[e] = v + br[e];
    }
    float mx = acc[0];
#pragma unroll
    for (int e = 1; e < E_; ++e) mx = fmaxf(mx, acc[e]);
    float sum = 0.f;
#pragma unroll
    for (int e = 0; e < E_; ++e) { acc[e] = __expf(acc[e] - mx); sum += acc[e]; }
    float inv = 1.f / sum;
    if (lane < E_) probs[(size_t)t * E_ + lane] = acc[lane] * inv;
}

// ---- fused dense-MoE MLP: 64-token tile per workgroup ---------------------------
__global__ __launch_bounds__(256) void moe_kernel(const float* __restrict__ x,
                                                  const __bf16* __restrict__ W1p,
                                                  const float* __restrict__ b1,
                                                  const __bf16* __restrict__ W2p,
                                                  const float* __restrict__ b2,
                                                  const float* __restrict__ probs,
                                                  float* __restrict__ out) {
    extern __shared__ char smem[];
    float*  ys = (float*)smem;                                                   // 64*772*4
    __bf16* xs = (__bf16*)(smem + 64 * YS_STRIDE * 4);                           // 64*776*2
    __bf16* hs = (__bf16*)(smem + 64 * YS_STRIDE * 4 + 64 * XS_STRIDE * 2);      // 64*136*2
    float*  ps = (float*)(smem + 64 * YS_STRIDE * 4 + 64 * XS_STRIDE * 2 + 64 * HS_STRIDE * 2);

    const int tid   = threadIdx.x;
    const int wave  = tid >> 5;
    const int lane  = tid & 31;
    const int laneN = lane & 15;
    const int laneH = lane >> 4;
    const int g     = wave >> 2;   // M-group: rows [32g, 32g+32)
    const int p     = wave & 3;    // N-group
    const int t0    = blockIdx.x * 64;

    if (tid < 128) ((float4*)ps)[tid] = ((const float4*)(probs + (size_t)t0 * E_))[tid];
    __syncthreads();

    // stage x tile as bf16 (reused for all experts / chunks)
    for (int i = tid; i < 64 * (D_ / 4); i += 256) {
        int r = i / (D_ / 4), c4 = i - r * (D_ / 4);
        float4 f = ((const float4*)(x + (size_t)(t0 + r) * D_))[c4];
        union { __bf16 h[4]; ushort4 u; } o;
        o.h[0] = (__bf16)f.x; o.h[1] = (__bf16)f.y;
        o.h[2] = (__bf16)f.z; o.h[3] = (__bf16)f.w;
        *(ushort4*)(xs + r * XS_STRIDE + c4 * 4) = o.u;
    }
    // ys := probs-weighted b2 (softmax sums to 1)
    {
        int r  = tid >> 2;
        int d0 = (tid & 3) * (D_ / 4);
        float pv[E_];
#pragma unroll
        for (int e = 0; e < E_; ++e) pv[e] = ps[r * E_ + e];
        for (int d = d0; d < d0 + (D_ / 4); ++d) {
            float v = 0.f;
#pragma unroll
            for (int e = 0; e < E_; ++e) v += pv[e] * b2[e * D_ + d];
            ys[r * YS_STRIDE + d] = v;
        }
    }
    __syncthreads();

    const f32x8 zero = {0.f, 0.f, 0.f, 0.f, 0.f, 0.f, 0.f, 0.f};

    for (int e = 0; e < E_; ++e) {
        const __bf16* W1e = W1p + (size_t)e * (H_ / 16) * (D_ / 32) * 512;
        const __bf16* W2e = W2p + (size_t)e * (D_ / 16) * (H_ / 32) * 512;
        for (int c = 0; c < H_ / HC; ++c) {
            const int hbase = c * HC;

            // ---- fc1: wave(g,p) -> M-tiles {2g,2g+1} x N-tiles {2p,2p+1} of HC ----
            f32x8 acc[2][2];
#pragma unroll
            for (int mi = 0; mi < 2; ++mi)
#pragma unroll
                for (int ni = 0; ni < 2; ++ni) acc[mi][ni] = zero;
            const __bf16* wb0 = W1e + ((size_t)(hbase / 16 + 2 * p) * (D_ / 32)) * 512 + lane * 16;
            const __bf16* wb1 = wb0 + (size_t)(D_ / 32) * 512;
            for (int k = 0; k < D_; k += 32) {
                const int ks = k >> 5;
                FragAB Bf[2];
                Bf[0].q[0] = *(const uint4*)(wb0 + ks * 512);
                Bf[0].q[1] = *(const uint4*)(wb0 + ks * 512 + 8);
                Bf[1].q[0] = *(const uint4*)(wb1 + ks * 512);
                Bf[1].q[1] = *(const uint4*)(wb1 + ks * 512 + 8);
                FragAB Af[2];
#pragma unroll
                for (int mi = 0; mi < 2; ++mi) {
                    const __bf16* ap = xs + (16 * (2 * g + mi) + laneN) * XS_STRIDE + k + 16 * laneH;
                    Af[mi].q[0] = *(const uint4*)(ap);
                    Af[mi].q[1] = *(const uint4*)(ap + 8);
                }
#pragma unroll
                for (int mi = 0; mi < 2; ++mi)
#pragma unroll
                    for (int ni = 0; ni < 2; ++ni)
                        acc[mi][ni] = WMMA_BF16(Af[mi], Bf[ni], acc[mi][ni]);
            }
            // bias + quickGELU + fold router prob, store bf16 hidden
#pragma unroll
            for (int ni = 0; ni < 2; ++ni) {
                const int hcol = hbase + (2 * p + ni) * 16 + laneN;
                const float bias = b1[e * H_ + hcol];
#pragma unroll
                for (int mi = 0; mi < 2; ++mi) {
#pragma unroll
                    for (int i = 0; i < 8; ++i) {
                        int row = 16 * (2 * g + mi) + 8 * laneH + i;
                        float v = acc[mi][ni][i] + bias;
                        float gg = v / (1.f + __expf(-1.702f * v));
                        gg *= ps[row * E_ + e];
                        hs[row * HS_STRIDE + (2 * p + ni) * 16 + laneN] = (__bf16)gg;
                    }
                }
            }
            __syncthreads();

            // ---- fc2: wave(g,p) -> M-tiles {2g,2g+1} x D-tiles [12p,12p+12) ----
            FragAB Ac[2][4];                       // hidden A-fragments cached in VGPRs
#pragma unroll
            for (int mi = 0; mi < 2; ++mi)
#pragma unroll
                for (int kk = 0; kk < 4; ++kk) {
                    const __bf16* ap = hs + (16 * (2 * g + mi) + laneN) * HS_STRIDE + kk * 32 + 16 * laneH;
                    Ac[mi][kk].q[0] = *(const uint4*)(ap);
                    Ac[mi][kk].q[1] = *(const uint4*)(ap + 8);
                }
#pragma unroll 1
            for (int j = 0; j < 12; ++j) {
                const int dt = 12 * p + j;
                const __bf16* wb = W2e + ((size_t)dt * (H_ / 32) + c * 4) * 512 + lane * 16;
                FragAB Bf[4];
#pragma unroll
                for (int kk = 0; kk < 4; ++kk) {
                    Bf[kk].q[0] = *(const uint4*)(wb + kk * 512);
                    Bf[kk].q[1] = *(const uint4*)(wb + kk * 512 + 8);
                }
#pragma unroll
                for (int mi = 0; mi < 2; ++mi) {
                    f32x8 cacc = zero;
#pragma unroll
                    for (int kk = 0; kk < 4; ++kk)
                        cacc = WMMA_BF16(Ac[mi][kk], Bf[kk], cacc);
                    float* yp = ys + (16 * (2 * g + mi) + 8 * laneH) * YS_STRIDE + dt * 16 + laneN;
#pragma unroll
                    for (int i = 0; i < 8; ++i)
                        (void)__hip_atomic_fetch_add(&yp[i * YS_STRIDE], cacc[i],
                                                     __ATOMIC_RELAXED, __HIP_MEMORY_SCOPE_WORKGROUP);
                }
            }
            __syncthreads();   // hs rewritten next chunk
        }
    }

    // coalesced f32 writeback
    {
        int r  = tid >> 2;
        int d0 = (tid & 3) * (D_ / 4);
        float* op = out + (size_t)(t0 + r) * D_;
        const float* yp = ys + r * YS_STRIDE;
        for (int d = d0; d < d0 + (D_ / 4); d += 4)
            *(float4*)(op + d) = *(const float4*)(yp + d);
    }
}

extern "C" void kernel_launch(void* const* d_in, const int* in_sizes, int n_in,
                              void* d_out, int out_size, void* d_ws, size_t ws_size,
                              hipStream_t stream) {
    const float* x  = (const float*)d_in[0];
    const float* W1 = (const float*)d_in[1];
    const float* b1 = (const float*)d_in[2];
    const float* W2 = (const float*)d_in[3];
    const float* b2 = (const float*)d_in[4];
    const float* Wr = (const float*)d_in[5];
    const float* br = (const float*)d_in[6];
    float* out = (float*)d_out;

    // ws: probs [T*8 f32] | W1 frags bf16 | W2 frags bf16  (~72.6 MB)
    char* ws = (char*)d_ws;
    float*  probs = (float*)ws;
    const size_t probsBytes = (size_t)T_ * E_ * sizeof(float);
    __bf16* W1p = (__bf16*)(ws + probsBytes);
    __bf16* W2p = (__bf16*)(ws + probsBytes + (size_t)E_ * H_ * D_ * 2);

    pack_frags_kernel<<<4608, 256, 0, stream>>>(W1, W1p, H_ / 16, D_ / 32);
    pack_frags_kernel<<<4608, 256, 0, stream>>>(W2, W2p, D_ / 16, H_ / 32);
    router_kernel<<<T_ / 8, 256, 0, stream>>>(x, Wr, br, probs);

    const size_t lds = (size_t)64 * YS_STRIDE * 4 + (size_t)64 * XS_STRIDE * 2 +
                       (size_t)64 * HS_STRIDE * 2 + (size_t)64 * E_ * 4;  // 316416 B
    moe_kernel<<<T_ / 64, 256, lds, stream>>>(x, W1p, b1, W2p, b2, probs, out);
}
